// IntEinsum_57286273794903
// MI455X (gfx1250) — matrix operand order repaired and checked by reading, not compiled, not run on hardware
//
#include <hip/hip_runtime.h>

// ---------------------------------------------------------------------------
// out[b,t,f] = sum_d X[b,t,d] * (W_int4[d,f] / scale[f])
//   M=1024, K=4096, N=16384  -> compute-bound GEMM (137 GFLOP vs ~6us HBM).
//
// Fast path (needs 144MB workspace):
//   prep_w : int4 W[K][N] -> f16 Wt[N][K] (transposed, exact in f16)
//   prep_x : fp32 X -> f16 Xhi + Xlo planes (Markidis split, ~fp32 accuracy)
//   gemm   : double-buffered LDS, global_load_async_to_lds_b128 staging
//            (ASYNCcnt), 16x v_wmma_f32_16x16x32_f16 per wave per K-step.
//            All fragments preloaded into distinct regs before the WMMA block
//            so s_wait_dscnt counts stagger instead of draining to 0.
// Fallback path: fused kernel (converts in-loop).
// ---------------------------------------------------------------------------

typedef _Float16 v16h __attribute__((ext_vector_type(16)));
typedef _Float16 v8h  __attribute__((ext_vector_type(8)));
typedef _Float16 v4h  __attribute__((ext_vector_type(4)));
typedef float    v8f  __attribute__((ext_vector_type(8)));

#define K_DIM   4096
#define N_DIM   16384
#define M_DIM   1024

#define BM 128
#define BN 128
#define BK 32
#define LDSK 40            // padded row stride in halves (80B, keeps 16B align)
#define PLANE (BM * LDSK)  // halves per LDS plane (one buffer)

// ---------------- CDNA5 async global->LDS copy (ASYNCcnt) -------------------
__device__ __forceinline__ void async_b128(unsigned int lds_byte_addr, const void* gaddr) {
    asm volatile("global_load_async_to_lds_b128 %0, %1, off"
                 :: "v"(lds_byte_addr), "v"(gaddr) : "memory");
}
__device__ __forceinline__ void wait_async0() {
    asm volatile("s_wait_asynccnt 0x0" ::: "memory");
}

// ======================= pre-pass: W int4 -> f16, transpose ================
__global__ __launch_bounds__(256) void prep_w(
    const signed char* __restrict__ Wq,   // [K][N] int4, 1 byte/elem
    _Float16* __restrict__ Wt)            // [N][K] f16
{
    __shared__ __align__(16) _Float16 sT[64][72];   // pad to 72 halves (144B)

    const int t  = threadIdx.x;
    const int k0 = blockIdx.x * 64;
    const int n0 = blockIdx.y * 64;

    {
        const int kr = t >> 2;             // 0..63
        const int nc = (t & 3) << 4;       // 0,16,32,48
        const int4* gp = reinterpret_cast<const int4*>(
            Wq + (size_t)(k0 + kr) * N_DIM + n0 + nc);
        int4 w = *gp;
        int d[4] = { w.x, w.y, w.z, w.w };
#pragma unroll
        for (int j = 0; j < 16; ++j) {
            int v = (d[j >> 2] << ((3 - (j & 3)) * 8)) >> 24;   // sign-extend
            sT[nc + j][kr] = (_Float16)v;
        }
    }
    __syncthreads();
    {
        const int nr = t >> 2;             // 0..63
        const int kc = (t & 3) << 4;       // 0,16,32,48
        v8h a = *(const v8h*)&sT[nr][kc];
        v8h b = *(const v8h*)&sT[nr][kc + 8];
        _Float16* op = Wt + (size_t)(n0 + nr) * K_DIM + k0 + kc;
        *(v8h*)op       = a;
        *((v8h*)op + 1) = b;
    }
}

// ======================= pre-pass: X fp32 -> hi/lo f16 =====================
__global__ __launch_bounds__(256) void prep_x(
    const float* __restrict__ X, _Float16* __restrict__ Xhi, _Float16* __restrict__ Xlo)
{
    const size_t i = ((size_t)blockIdx.x * 256 + threadIdx.x) * 4;
    float4 f = *reinterpret_cast<const float4*>(X + i);
    float xs[4] = { f.x, f.y, f.z, f.w };
    v4h hi, lo;
#pragma unroll
    for (int j = 0; j < 4; ++j) {
        _Float16 h = (_Float16)xs[j];
        hi[j] = h;
        lo[j] = (_Float16)(xs[j] - (float)h);
    }
    *(v4h*)(Xhi + i) = hi;
    *(v4h*)(Xlo + i) = lo;
}

// ---------------- fragment helpers -----------------------------------------
__device__ __forceinline__ v16h load_frag32(const _Float16* p, int second_off) {
    v8h a = *(const v8h*)p;
    v8h b = *(const v8h*)(p + second_off);
    return __builtin_shufflevector(a, b, 0,1,2,3,4,5,6,7,8,9,10,11,12,13,14,15);
}

// ======================= main GEMM (fast path) =============================
__global__ __launch_bounds__(256) void gemm_f16_async(
    const _Float16* __restrict__ Xhi,   // [M][K]
    const _Float16* __restrict__ Xlo,   // [M][K]
    const _Float16* __restrict__ Wt,    // [N][K]
    const float* __restrict__ scale,    // [N]
    float* __restrict__ Out)            // [M][N]
{
    __shared__ __align__(16) _Float16 sAhi[2 * PLANE];
    __shared__ __align__(16) _Float16 sAlo[2 * PLANE];
    __shared__ __align__(16) _Float16 sB  [2 * PLANE];

    const int tid  = threadIdx.x;
    const int lane = tid & 31;
    const int wid  = tid >> 5;
    const int wm   = wid >> 2;          // 0..1
    const int wn   = wid & 3;           // 0..3
    const int n0 = blockIdx.x * BN;
    const int m0 = blockIdx.y * BM;
    const int lm = lane & 15;
    const int lh = lane >> 4;

    v8f acc[4][2];
#pragma unroll
    for (int mi = 0; mi < 4; ++mi)
#pragma unroll
        for (int ni = 0; ni < 2; ++ni)
            acc[mi][ni] = (v8f){0.f,0.f,0.f,0.f,0.f,0.f,0.f,0.f};

    // staging: each thread copies one 32B (16-half) chunk per plane per K-step
    const int crow = tid >> 1;             // 0..127 (A row / B col)
    const int coff = (tid & 1) << 4;       // half offset 0 or 16
    const unsigned int planeByte = (unsigned int)((crow * LDSK + coff) * 2);
    const unsigned int ldsAhi0 = (unsigned int)(size_t)sAhi + planeByte;
    const unsigned int ldsAlo0 = (unsigned int)(size_t)sAlo + planeByte;
    const unsigned int ldsB0   = (unsigned int)(size_t)sB   + planeByte;
    const _Float16* gAhi = Xhi + (size_t)(m0 + crow) * K_DIM + coff;
    const _Float16* gAlo = Xlo + (size_t)(m0 + crow) * K_DIM + coff;
    const _Float16* gB   = Wt  + (size_t)(n0 + crow) * K_DIM + coff;

    auto issue = [&](int buf, int k0h) {
        const unsigned int bo = (unsigned int)buf * (PLANE * 2);  // bytes
        async_b128(ldsAhi0 + bo,      gAhi + k0h);
        async_b128(ldsAhi0 + bo + 16, gAhi + k0h + 8);
        async_b128(ldsAlo0 + bo,      gAlo + k0h);
        async_b128(ldsAlo0 + bo + 16, gAlo + k0h + 8);
        async_b128(ldsB0   + bo,      gB   + k0h);
        async_b128(ldsB0   + bo + 16, gB   + k0h + 8);
    };

    issue(0, 0);
    int buf = 0;
    for (int kt = 0; kt < K_DIM / BK; ++kt) {
        wait_async0();          // current buffer's copies (this wave's) done
        __syncthreads();        // -> all waves' copies done; prev compute done
        if (kt + 1 < K_DIM / BK) issue(buf ^ 1, (kt + 1) * BK);

        const int hb = buf * PLANE;
        const int kb = lh << 3;     // 0 or 8

        // ---- preload ALL fragments (distinct regs; one big ds clause) ----
        v16h bf[2], ahi[4], alo[4];
#pragma unroll
        for (int ni = 0; ni < 2; ++ni)
            bf[ni] = load_frag32(&sB[hb + (wn * 32 + ni * 16 + lm) * LDSK + (lh << 3)], 16);
#pragma unroll
        for (int mi = 0; mi < 4; ++mi) {
            const int row = wm * 64 + mi * 16 + lm;
            ahi[mi] = load_frag32(&sAhi[hb + row * LDSK + kb], 16);
            alo[mi] = load_frag32(&sAlo[hb + row * LDSK + kb], 16);
        }

        // ---- 16 WMMAs per wave per K-step ----
#pragma unroll
        for (int mi = 0; mi < 4; ++mi)
#pragma unroll
            for (int ni = 0; ni < 2; ++ni) {
                acc[mi][ni] = __builtin_amdgcn_wmma_f32_16x16x32_f16(
                    false, ahi[mi], false, bf[ni], (short)0, acc[mi][ni], false, false);
                acc[mi][ni] = __builtin_amdgcn_wmma_f32_16x16x32_f16(
                    false, alo[mi], false, bf[ni], (short)0, acc[mi][ni], false, false);
            }
        buf ^= 1;
    }

    // epilogue: out = acc * (1/scale[n]);  C layout: VGPR r -> row r+8*lh, col lm
#pragma unroll
    for (int mi = 0; mi < 4; ++mi) {
#pragma unroll
        for (int ni = 0; ni < 2; ++ni) {
            const int col  = n0 + wn * 32 + ni * 16 + lm;
            const int rowb = m0 + wm * 64 + mi * 16 + (lh << 3);
            const float s  = 1.0f / scale[col];
            float* op = Out + (size_t)rowb * N_DIM + col;
            v8f a = acc[mi][ni];
#pragma unroll
            for (int r = 0; r < 8; ++r)
                op[(size_t)r * N_DIM] = a[r] * s;
        }
    }
}

// ======================= fallback: fused ===================================
__global__ __launch_bounds__(256) void int4_gemm_wmma_kernel(
    const float* __restrict__ X, const signed char* __restrict__ Wq,
    const float* __restrict__ scale, float* __restrict__ Out)
{
    __shared__ __align__(16) _Float16 sAhi[PLANE];
    __shared__ __align__(16) _Float16 sAlo[PLANE];
    __shared__ __align__(16) _Float16 sB  [PLANE];

    const int tid  = threadIdx.x;
    const int lane = tid & 31;
    const int wid  = tid >> 5;
    const int wm   = wid >> 2;
    const int wn   = wid & 3;
    const int n0 = blockIdx.x * BN;
    const int m0 = blockIdx.y * BM;
    const int ar = tid >> 1;
    const int ak = (tid & 1) << 4;
    const int bk = tid >> 3;
    const int bc = (tid & 7) << 4;
    const int lm = lane & 15;
    const int lh = lane >> 4;

    v8f acc[4][2];
#pragma unroll
    for (int mi = 0; mi < 4; ++mi)
#pragma unroll
        for (int ni = 0; ni < 2; ++ni)
            acc[mi][ni] = (v8f){0.f,0.f,0.f,0.f,0.f,0.f,0.f,0.f};

    const float*       agp = X  + (m0 + ar) * K_DIM + ak;
    const signed char* bgp = Wq + bk * N_DIM + n0 + bc;

    for (int kt = 0; kt < K_DIM / BK; ++kt) {
        const int k0 = kt * BK;
        {
            const float4* gp = reinterpret_cast<const float4*>(agp + k0);
            float4 f[4];
#pragma unroll
            for (int i = 0; i < 4; ++i) f[i] = gp[i];
            float xv[16] = { f[0].x,f[0].y,f[0].z,f[0].w, f[1].x,f[1].y,f[1].z,f[1].w,
                             f[2].x,f[2].y,f[2].z,f[2].w, f[3].x,f[3].y,f[3].z,f[3].w };
            v8h hi0, hi1, lo0, lo1;
#pragma unroll
            for (int i = 0; i < 8; ++i) {
                _Float16 h = (_Float16)xv[i];
                hi0[i] = h; lo0[i] = (_Float16)(xv[i] - (float)h);
                _Float16 h2 = (_Float16)xv[i + 8];
                hi1[i] = h2; lo1[i] = (_Float16)(xv[i + 8] - (float)h2);
            }
            *(v8h*)&sAhi[ar * LDSK + ak]     = hi0;
            *(v8h*)&sAhi[ar * LDSK + ak + 8] = hi1;
            *(v8h*)&sAlo[ar * LDSK + ak]     = lo0;
            *(v8h*)&sAlo[ar * LDSK + ak + 8] = lo1;
        }
        {
            const int4* gp = reinterpret_cast<const int4*>(bgp + k0 * N_DIM);
            int4 w = *gp;
            int d[4] = { w.x, w.y, w.z, w.w };
#pragma unroll
            for (int j = 0; j < 16; ++j) {
                int v = (d[j >> 2] << ((3 - (j & 3)) * 8)) >> 24;
                sB[(bc + j) * LDSK + bk] = (_Float16)v;
            }
        }
        if (kt + 1 < K_DIM / BK) {
            __builtin_prefetch(agp + k0 + BK, 0, 3);
            __builtin_prefetch(bgp + (k0 + BK) * N_DIM, 0, 3);
        }
        __syncthreads();

        const int kb = lh << 3;
        v16h bf[2], ahi[4], alo[4];
#pragma unroll
        for (int ni = 0; ni < 2; ++ni)
            bf[ni] = load_frag32(&sB[(wn * 32 + ni * 16 + lm) * LDSK + (lh << 3)], 16);
#pragma unroll
        for (int mi = 0; mi < 4; ++mi) {
            const int row = wm * 64 + mi * 16 + lm;
            ahi[mi] = load_frag32(&sAhi[row * LDSK + kb], 16);
            alo[mi] = load_frag32(&sAlo[row * LDSK + kb], 16);
        }
#pragma unroll
        for (int mi = 0; mi < 4; ++mi)
#pragma unroll
            for (int ni = 0; ni < 2; ++ni) {
                acc[mi][ni] = __builtin_amdgcn_wmma_f32_16x16x32_f16(
                    false, ahi[mi], false, bf[ni], (short)0, acc[mi][ni], false, false);
                acc[mi][ni] = __builtin_amdgcn_wmma_f32_16x16x32_f16(
                    false, alo[mi], false, bf[ni], (short)0, acc[mi][ni], false, false);
            }
        __syncthreads();
    }
#pragma unroll
    for (int mi = 0; mi < 4; ++mi) {
#pragma unroll
        for (int ni = 0; ni < 2; ++ni) {
            const int col  = n0 + wn * 32 + ni * 16 + lm;
            const int rowb = m0 + wm * 64 + mi * 16 + (lh << 3);
            const float s  = 1.0f / scale[col];
            float* op = Out + (size_t)rowb * N_DIM + col;
            v8f a = acc[mi][ni];
#pragma unroll
            for (int r = 0; r < 8; ++r)
                op[(size_t)r * N_DIM] = a[r] * s;
        }
    }
}

// ======================= launcher ==========================================
extern "C" void kernel_launch(void* const* d_in, const int* in_sizes, int n_in,
                              void* d_out, int out_size, void* d_ws, size_t ws_size,
                              hipStream_t stream) {
    const float*       X     = (const float*)d_in[0];
    const signed char* Wq    = (const signed char*)d_in[1];  // ml_dtypes int4: 1B/elem
    const float*       scale = (const float*)d_in[2];
    float*             Out   = (float*)d_out;

    const size_t wtBytes = (size_t)N_DIM * K_DIM * sizeof(_Float16);   // 128 MB
    const size_t xpBytes = (size_t)M_DIM * K_DIM * sizeof(_Float16);   //   8 MB
    const size_t need    = wtBytes + 2 * xpBytes;                      // 144 MB

    if (ws_size >= need && d_ws != nullptr) {
        _Float16* Wt  = (_Float16*)d_ws;
        _Float16* Xhi = (_Float16*)((char*)d_ws + wtBytes);
        _Float16* Xlo = Xhi + (size_t)M_DIM * K_DIM;

        prep_w<<<dim3(K_DIM / 64, N_DIM / 64), 256, 0, stream>>>(Wq, Wt);
        prep_x<<<dim3((M_DIM * K_DIM) / (256 * 4)), 256, 0, stream>>>(X, Xhi, Xlo);
        gemm_f16_async<<<dim3(N_DIM / BN, M_DIM / BM), 256, 0, stream>>>(
            Xhi, Xlo, Wt, scale, Out);
    } else {
        int4_gemm_wmma_kernel<<<dim3(N_DIM / BN, M_DIM / BM), 256, 0, stream>>>(
            X, Wq, scale, Out);
    }
}
